// NIC_Decoder_59459527246005
// MI455X (gfx1250) — matrix operand order repaired and checked by reading, not compiled, not run on hardware
//
#include <hip/hip_runtime.h>
#include <hip/hip_bf16.h>

// ---------------- problem constants ----------------
#define B_    64
#define T1_   41
#define E_    512
#define H_    512
#define V_    10000
#define FOURH 2048
#define BT    (B_ * T1_)      // 2624
#define MT_   (BT / 16)       // 164 M-tiles

// ---------------- WMMA types ----------------
typedef __attribute__((ext_vector_type(16))) __bf16 v16bf;
typedef __attribute__((ext_vector_type(8)))  float  v8f;

struct alignas(16) Q4 { unsigned x, y, z, w; };
union Frag { Q4 q[2]; v16bf v; };

static __device__ __forceinline__ v8f wmma_bf16(v16bf a, v16bf b, v8f c) {
    return __builtin_amdgcn_wmma_f32_16x16x32_bf16(false, a, false, b,
                                                   (short)0, c, false, false);
}
#define SCHED_FENCE() __builtin_amdgcn_sched_barrier(0)

// ---------------- helpers ----------------
static __device__ __forceinline__ unsigned f2bf(float f) {  // RNE f32->bf16
    unsigned u = __float_as_uint(f);
    return (u + 0x7FFFu + ((u >> 16) & 1u)) >> 16;
}
static __device__ __forceinline__ float sigm(float x) {
    return 1.0f / (1.0f + __expf(-x));
}
static __device__ __forceinline__ unsigned rotl(unsigned v, int d) {
    return (v << d) | (v >> (32 - d));
}

// Exact JAX threefry2x32 (5 groups of 4 rounds, key injections)
static __device__ void threefry2x32(unsigned k0, unsigned k1,
                                    unsigned x0, unsigned x1,
                                    unsigned& o0, unsigned& o1) {
    unsigned k2 = k0 ^ k1 ^ 0x1BD11BDAu;
    const int R0[4] = {13, 15, 26, 6};
    const int R1[4] = {17, 29, 16, 24};
    x0 += k0; x1 += k1;
#pragma unroll
    for (int i = 0; i < 4; i++) { x0 += x1; x1 = rotl(x1, R0[i]); x1 ^= x0; }
    x0 += k1; x1 += k2 + 1u;
#pragma unroll
    for (int i = 0; i < 4; i++) { x0 += x1; x1 = rotl(x1, R1[i]); x1 ^= x0; }
    x0 += k2; x1 += k0 + 2u;
#pragma unroll
    for (int i = 0; i < 4; i++) { x0 += x1; x1 = rotl(x1, R0[i]); x1 ^= x0; }
    x0 += k0; x1 += k1 + 3u;
#pragma unroll
    for (int i = 0; i < 4; i++) { x0 += x1; x1 = rotl(x1, R1[i]); x1 ^= x0; }
    x0 += k1; x1 += k2 + 4u;
#pragma unroll
    for (int i = 0; i < 4; i++) { x0 += x1; x1 = rotl(x1, R0[i]); x1 ^= x0; }
    x0 += k2; x1 += k0 + 5u;
    o0 = x0; o1 = x1;
}

static __device__ __forceinline__ float bern2(unsigned bits) {
    // jax uniform: bitcast((bits>>9)|0x3f800000) - 1 ; keep if < 0.5 ; /0.5
    float u = __uint_as_float((bits >> 9) | 0x3f800000u) - 1.0f;
    return (u < 0.5f) ? 2.0f : 0.0f;
}

// ---------------- kernel 1: pack f32 [K x N] -> bf16 WMMA B-fragment order ----
// dst dword id = ((nt*K32 + kc)*32 + lane)*8 + r
// col = nt*16+(lane&15), k = kc*32 + 8*(lane>>4) + (r<4 ? 2r : 16+2(r-4)) {,+1}
__global__ void __launch_bounds__(256)
pack_b_kernel(const float* __restrict__ src, unsigned* __restrict__ dst,
              int K32, int Ntiles, int N) {
    int id = blockIdx.x * 256 + threadIdx.x;
    int total = Ntiles * K32 * 256;
    if (id >= total) return;
    int r    = id & 7;
    int lane = (id >> 3) & 31;
    int frag = id >> 8;
    int kc   = frag % K32;
    int nt   = frag / K32;
    int col  = nt * 16 + (lane & 15);
    int k    = kc * 32 + ((lane < 16) ? 0 : 8) + ((r < 4) ? 2 * r : 16 + 2 * (r - 4));
    unsigned lo = f2bf(src[(size_t)k * N + col]);
    unsigned hi = f2bf(src[(size_t)(k + 1) * N + col]);
    dst[id] = lo | (hi << 16);
}

// ---------------- kernel 2: dropout masks (threefry, key 42) ----------------
#define WPAIR 16384     // word mask (64,1,512)  -> 32768 elems
#define RPAIR 16384     // rec  mask (64,512)    -> 32768 elems
#define FPAIR 671744    // final mask (64,41,512)-> 1343488 elems
__global__ void __launch_bounds__(256)
masks_kernel(float* __restrict__ wmask, float* __restrict__ rmask,
             float* __restrict__ fmask) {
    int id = blockIdx.x * 256 + threadIdx.x;
    if (id >= WPAIR + RPAIR + FPAIR) return;
    // dk = split(key(42), 3): bits over counts 0..5 -> pairs (0,3),(1,4),(2,5)
    unsigned a0, a1, b0, b1, c0, c1;
    threefry2x32(0u, 42u, 0u, 3u, a0, a1);
    threefry2x32(0u, 42u, 1u, 4u, b0, b1);
    threefry2x32(0u, 42u, 2u, 5u, c0, c1);
    // reshape(3,2): dk0=(a0,b0) dk1=(c0,a1) dk2=(b1,c1)
    unsigned o0, o1;
    if (id < WPAIR) {
        unsigned i = id;
        threefry2x32(a0, b0, i, i + WPAIR, o0, o1);
        wmask[i] = bern2(o0); wmask[i + WPAIR] = bern2(o1);
    } else if (id < WPAIR + RPAIR) {
        unsigned i = id - WPAIR;
        threefry2x32(c0, a1, i, i + RPAIR, o0, o1);
        rmask[i] = bern2(o0); rmask[i + RPAIR] = bern2(o1);
    } else {
        unsigned i = id - WPAIR - RPAIR;
        threefry2x32(b1, c1, i, i + FPAIR, o0, o1);
        fmask[i] = bern2(o0); fmask[i + FPAIR] = bern2(o1);
    }
}

// ---------------- kernel 3: build x_seq (bf16) -------------------------------
__global__ void __launch_bounds__(256)
build_x_kernel(const float* __restrict__ img, const int* __restrict__ tgt,
               const float* __restrict__ emb, const float* __restrict__ wmask,
               unsigned short* __restrict__ xs) {
    int id = blockIdx.x * 256 + threadIdx.x;       // (B,T1,E) flat
    if (id >= B_ * T1_ * E_) return;
    int e  = id & 511;
    int bt = id >> 9;
    int t  = bt % T1_;
    int b  = bt / T1_;
    float v;
    if (t == 0) v = img[b * E_ + e];
    else        v = emb[(size_t)tgt[b * T1_ + t - 1] * E_ + e] * wmask[b * E_ + e];
    xs[id] = (unsigned short)f2bf(v);
}

// ---------------- kernel 4: zero h/c state ----------------------------------
__global__ void __launch_bounds__(256)
zero_state_kernel(float* __restrict__ c, unsigned* __restrict__ hm_u) {
    int id = blockIdx.x * 256 + threadIdx.x;
    if (id < B_ * H_) c[id] = 0.0f;
    if (id < (B_ * H_) / 2) hm_u[id] = 0u;
}

// ---------------- kernel 5: generic bf16 WMMA GEMM (+bias) ------------------
// C[M,N] = A[M,K](bf16 row-major) @ Bpk(packed) + bias[N]
// One wave computes a 16 x (16*NT) strip. NO runtime guards: grids are exact,
// so EXEC is all-1s at every WMMA. Depth-2 software pipeline: two persistent
// B buffer sets; sched_barrier(0) fences lock WMMA/refill stage order so each
// buffer's load->use distance is two full WMMA groups (counted waits, not 0).
template <int NT, int K32C>
__global__ void __launch_bounds__(128)
gemm_bf16_kernel(const unsigned short* __restrict__ A,
                 const unsigned* __restrict__ Bpk,
                 const float* __restrict__ bias, float* __restrict__ C,
                 int ntPerRow, int K, int N) {
    int wid  = (blockIdx.x * 128 + threadIdx.x) >> 5;
    int lane = threadIdx.x & 31;
    int mt   = wid / ntPerRow;
    int nt0  = (wid % ntPerRow) * NT;
    int rl   = lane & 15;
    int half = lane >> 4;

    v8f acc[NT];
#pragma unroll
    for (int j = 0; j < NT; j++) {
        float bv = bias[(nt0 + j) * 16 + rl];
#pragma unroll
        for (int r = 0; r < 8; r++) acc[j][r] = bv;
    }
    const unsigned short* Arow = A + (size_t)(mt * 16 + rl) * K + half * 8;
    const unsigned* Bbase = Bpk + (size_t)nt0 * K32C * 256 + lane * 8;

    auto loadB = [&](Frag* buf, int kcv) {
#pragma unroll
        for (int j = 0; j < NT; j++) {
            const Q4* bp = (const Q4*)(Bbase + ((size_t)j * K32C + kcv) * 256);
            buf[j].q[0] = bp[0]; buf[j].q[1] = bp[1];
        }
    };
    auto loadA = [&](Frag& af, int kcv) {
        af.q[0] = *(const Q4*)(Arow + kcv * 32);
        af.q[1] = *(const Q4*)(Arow + kcv * 32 + 16);
    };

    Frag bnA[NT], bnB[NT];
    loadB(bnA, 0);
    loadB(bnB, 1);
#pragma unroll
    for (int kc = 0; kc < K32C; kc += 2) {
        Frag af0; loadA(af0, kc);
        SCHED_FENCE();
#pragma unroll
        for (int j = 0; j < NT; j++) acc[j] = wmma_bf16(af0.v, bnA[j].v, acc[j]);
        SCHED_FENCE();
        if (kc + 2 < K32C) loadB(bnA, kc + 2);
        Frag af1; loadA(af1, kc + 1);
        SCHED_FENCE();
#pragma unroll
        for (int j = 0; j < NT; j++) acc[j] = wmma_bf16(af1.v, bnB[j].v, acc[j]);
        SCHED_FENCE();
        if (kc + 3 < K32C) loadB(bnB, kc + 3);
    }
#pragma unroll
    for (int j = 0; j < NT; j++) {
#pragma unroll
        for (int r = 0; r < 8; r++) {
            int row = mt * 16 + half * 8 + r;
            C[(size_t)row * N + (nt0 + j) * 16 + rl] = acc[j][r];
        }
    }
}

// ---------------- kernel 6: fused LSTM step ---------------------------------
// 4 blocks x 512 thr (16 waves -> 4/SIMD); block handles 16 batch rows. Each
// wave computes matched i/f/g/o WMMA tiles for the same 16 hidden cols
// (2 hidden tiles/wave), then the cell update in registers. U fragments use
// the same depth-2 pipeline with sched fences as the generic GEMM.
__global__ void __launch_bounds__(512)
lstm_step_kernel(const float* __restrict__ xz, const unsigned* __restrict__ Upk,
                 unsigned short* __restrict__ hm, float* __restrict__ cst,
                 const float* __restrict__ rmask, const float* __restrict__ fmask,
                 unsigned short* __restrict__ hout, int t) {
    __shared__ unsigned short hm_s[16 * H_];        // 16 KB
    int b0 = blockIdx.x * 16;
    {   // stage h*rec_mask (bf16) rows b0..b0+15 into LDS
        const Q4* src = (const Q4*)(hm + (size_t)b0 * H_);
        Q4* dst = (Q4*)hm_s;
        for (int i = threadIdx.x; i < 16 * H_ / 8; i += 512) dst[i] = src[i];
    }
    __syncthreads();
    int w    = threadIdx.x >> 5;                    // 0..15
    int lane = threadIdx.x & 31;
    int rl   = lane & 15;
    int half = lane >> 4;

    for (int hbi = 0; hbi < 2; hbi++) {
        int hb = w * 2 + hbi;                       // hidden tile 0..31
        int n0 = hb * 16;
        v8f acc[4];
#pragma unroll
        for (int g = 0; g < 4; g++)
#pragma unroll
            for (int r = 0; r < 8; r++) {
                int bb = b0 + half * 8 + r;
                acc[g][r] = xz[((size_t)bb * T1_ + t) * FOURH + g * H_ + n0 + rl];
            }
        const unsigned short* As = hm_s + rl * H_ + half * 8;
        const unsigned* Ubase = Upk + (size_t)hb * 16 * 256 + lane * 8;

        auto loadB = [&](Frag* buf, int kcv) {
#pragma unroll
            for (int g = 0; g < 4; g++) {
                const Q4* bp = (const Q4*)(Ubase + ((size_t)g * 32 * 16 + kcv) * 256);
                buf[g].q[0] = bp[0]; buf[g].q[1] = bp[1];
            }
        };
        auto loadA = [&](Frag& af, int kcv) {
            af.q[0] = *(const Q4*)(As + kcv * 32);
            af.q[1] = *(const Q4*)(As + kcv * 32 + 16);
        };

        Frag bnA[4], bnB[4];
        loadB(bnA, 0);
        loadB(bnB, 1);
#pragma unroll
        for (int kc = 0; kc < 16; kc += 2) {
            Frag af0; loadA(af0, kc);
            SCHED_FENCE();
#pragma unroll
            for (int g = 0; g < 4; g++) acc[g] = wmma_bf16(af0.v, bnA[g].v, acc[g]);
            SCHED_FENCE();
            if (kc + 2 < 16) loadB(bnA, kc + 2);
            Frag af1; loadA(af1, kc + 1);
            SCHED_FENCE();
#pragma unroll
            for (int g = 0; g < 4; g++) acc[g] = wmma_bf16(af1.v, bnB[g].v, acc[g]);
            SCHED_FENCE();
            if (kc + 3 < 16) loadB(bnB, kc + 3);
        }
        // cell update directly on accumulator layout
#pragma unroll
        for (int r = 0; r < 8; r++) {
            int bb = b0 + half * 8 + r;
            int hx = n0 + rl;
            float iv = sigm(acc[0][r]);
            float fv = sigm(acc[1][r]);
            float gv = tanhf(acc[2][r]);
            float ov = sigm(acc[3][r]);
            size_t ci = (size_t)bb * H_ + hx;
            float cn = fv * cst[ci] + iv * gv;
            cst[ci] = cn;
            float hn = ov * tanhf(cn);
            hm[ci] = (unsigned short)f2bf(hn * rmask[ci]);
            size_t oi = ((size_t)bb * T1_ + t) * H_ + hx;
            hout[oi] = (unsigned short)f2bf(hn * fmask[oi]);
        }
    }
}

// ---------------- host launcher ---------------------------------------------
extern "C" void kernel_launch(void* const* d_in, const int* in_sizes, int n_in,
                              void* d_out, int out_size, void* d_ws, size_t ws_size,
                              hipStream_t stream) {
    (void)in_sizes; (void)n_in; (void)out_size; (void)ws_size;
    const float* img = (const float*)d_in[1];   // images_emb (B,E)
    const int*   tgt = (const int*)d_in[2];     // targets (B,T1)
    const float* emb = (const float*)d_in[3];   // emb_table (V,E)
    const float* W   = (const float*)d_in[4];   // (E,4H)
    const float* U   = (const float*)d_in[5];   // (H,4H)
    const float* b   = (const float*)d_in[6];   // (4H,)
    const float* Wo  = (const float*)d_in[7];   // (H,V)
    const float* bo  = (const float*)d_in[8];   // (V,)
    float* out = (float*)d_out;                 // (B,T1,V)

    size_t off = 0;
    char* base = (char*)d_ws;
    auto take = [&](size_t bytes) -> void* {
        void* p = base + off; off += (bytes + 255) & ~(size_t)255; return p;
    };
    unsigned*       Wpk   = (unsigned*)take((size_t)128 * 16 * 256 * 4);   // 2 MB
    unsigned*       Upk   = (unsigned*)take((size_t)128 * 16 * 256 * 4);   // 2 MB
    unsigned*       Wopk  = (unsigned*)take((size_t)625 * 16 * 256 * 4);   // 10.24 MB
    unsigned short* xs    = (unsigned short*)take((size_t)BT * E_ * 2);    // 2.7 MB
    float*          xz    = (float*)take((size_t)BT * FOURH * 4);          // 21.5 MB
    unsigned short* hout  = (unsigned short*)take((size_t)BT * H_ * 2);    // 2.7 MB
    float*          wmask = (float*)take((size_t)B_ * E_ * 4);
    float*          rmask = (float*)take((size_t)B_ * H_ * 4);
    float*          fmask = (float*)take((size_t)B_ * T1_ * H_ * 4);       // 5.4 MB
    float*          cst   = (float*)take((size_t)B_ * H_ * 4);
    unsigned short* hm    = (unsigned short*)take((size_t)B_ * H_ * 2);

    // prep: pack weights to bf16 WMMA-B layout
    pack_b_kernel<<<2048, 256, 0, stream>>>(W,  Wpk,  16, 128, FOURH);
    pack_b_kernel<<<2048, 256, 0, stream>>>(U,  Upk,  16, 128, FOURH);
    pack_b_kernel<<<10000, 256, 0, stream>>>(Wo, Wopk, 16, 625, V_);
    // prep: dropout masks + input sequence + state init
    masks_kernel<<<(WPAIR + RPAIR + FPAIR) / 256, 256, 0, stream>>>(wmask, rmask, fmask);
    build_x_kernel<<<(B_ * T1_ * E_) / 256, 256, 0, stream>>>(img, tgt, emb, wmask, xs);
    zero_state_kernel<<<(B_ * H_) / 256, 256, 0, stream>>>(cst, (unsigned*)hm);

    // GEMM1: xz = x_seq @ W + b   (M=2624, N=2048, K=512)
    // 128 N-tiles = 32 groups of 4; waves = 164*32 = 5248 = 1312 blocks * 4
    gemm_bf16_kernel<4, 16><<<1312, 128, 0, stream>>>(xs, Wpk, b, xz,
                                                      32, E_, FOURH);
    // recurrence: 41 fused step kernels
    for (int t = 0; t < T1_; t++)
        lstm_step_kernel<<<B_ / 16, 512, 0, stream>>>(xz, Upk, hm, cst,
                                                      rmask, fmask, hout, t);
    // GEMM3: out = (h*final_mask) @ Wo + bo   (M=2624, N=10000, K=512)
    // 625 N-tiles = 125 groups of 5; waves = 164*125 = 20500 = 5125 blocks * 4
    gemm_bf16_kernel<5, 16><<<5125, 128, 0, stream>>>(hout, Wopk, bo, out,
                                                      125, H_, V_);
}
